// RNNClassifier_47175920779363
// MI455X (gfx1250) — matrix operand order repaired and checked by reading, not compiled, not run on hardware
//
#include <hip/hip_runtime.h>
#include <hip/hip_bf16.h>
#include <cstdint>

#define VOCAB 50257
#define EMB   256
#define HID   512
#define TT    512
#define BB    64
#define G4    (4 * HID)   // 2048 gate outputs per step

// LDS slab geometry (bytes), padded rows for bank-conflict-free ds_load_b128
#define RPITCH 1040              // 512 bf16 + 8 bf16 pad  (65 uint4)
#define IPITCH 528               // 256 bf16 + 8 bf16 pad  (33 uint4)
#define RSLAB  (64 * RPITCH)     // 66560 B
#define ISLAB  (64 * IPITCH)     // 33792 B
#define SMEMQ  ((RSLAB + ISLAB) / 16)   // uint4 elements (6272 -> ~98 KB)

typedef __attribute__((ext_vector_type(16))) __bf16 v16bf;
typedef __attribute__((ext_vector_type(8)))  float  v8f;

union ABf { v16bf v; uint4 q[2]; };

__device__ __forceinline__ float fast_sigmoid(float x) {
    return 1.0f / (1.0f + __expf(-x));
}
__device__ __forceinline__ float fast_tanh(float x) {
    return 2.0f / (1.0f + __expf(-2.0f * x)) - 1.0f;
}

// One-time per-call prep: bf16 weight conversion, bias folding, h/c zero-init.
__global__ __launch_bounds__(256) void prep_kernel(
    const float* __restrict__ Wih, const float* __restrict__ Whh,
    const float* __restrict__ bih, const float* __restrict__ bhh,
    __bf16* __restrict__ WihB, __bf16* __restrict__ WhhB,
    float* __restrict__ bsum,
    __bf16* __restrict__ hbf, float* __restrict__ hf, float* __restrict__ cf)
{
    const int i = blockIdx.x * blockDim.x + threadIdx.x;
    if (i < G4 * HID) WhhB[i] = (__bf16)Whh[i];
    if (i < G4 * EMB) WihB[i] = (__bf16)Wih[i];
    if (i < G4)       bsum[i] = bih[i] + bhh[i];
    if (i < BB * HID) { hf[i] = 0.0f; cf[i] = 0.0f; hbf[i] = (__bf16)0.0f; }
}

// One LSTM timestep. 32 blocks (N-tiles over HID) x 128 threads (4 waves = M-tiles over B).
// Block stages its weight slab into LDS via async-to-LDS copies, then each wave
// accumulates its 4 gate tiles (i,f,g,o) with bf16 WMMA (B operand from LDS).
__global__ __launch_bounds__(128) void lstm_step_kernel(
    const int* __restrict__ x, const int* __restrict__ x_len,
    const float* __restrict__ emb,
    const __bf16* __restrict__ WihB, const __bf16* __restrict__ WhhB,
    const float* __restrict__ bsum,
    __bf16* __restrict__ hbf, float* __restrict__ hf, float* __restrict__ cf,
    int t)
{
    __shared__ __align__(16) uint4 smemQ[SMEMQ];

    const int tid  = threadIdx.x;
    const int lane = threadIdx.x & 31;
    const int wave = threadIdx.x >> 5;
    const int m0 = wave * 16;            // batch-tile origin
    const int n0 = blockIdx.x * 16;      // hidden-tile origin
    const int lm = lane & 15;
    const int lh = lane >> 4;

    // ---- async stage of this block's weight slab into LDS (ASYNCcnt path) ----
    {
        const uint32_t lds_base = (uint32_t)(uintptr_t)(void*)smemQ;
        // recurrent slab: 64 rows x 1024 B payload = 4096 x 16 B chunks
        const char* gW = (const char*)WhhB;
        for (int j = 0; j < 32; ++j) {
            const int c   = tid + j * 128;     // chunk id
            const int row = c >> 6;            // 0..63 (g*16 + rr)
            const int ci  = c & 63;            // 16 B chunk within row
            const int g   = row >> 4, rr = row & 15;
            const char* gp = gW + ((size_t)(g * HID + n0 + rr) * HID) * 2 + ci * 16;
            const uint32_t lo = lds_base + (uint32_t)(row * RPITCH + ci * 16);
            asm volatile("global_load_async_to_lds_b128 %0, %1, off"
                         :: "v"(lo), "v"(gp) : "memory");
        }
        // input slab: 64 rows x 512 B payload = 2048 x 16 B chunks
        const char* gI = (const char*)WihB;
        for (int j = 0; j < 16; ++j) {
            const int c   = tid + j * 128;
            const int row = c >> 5;            // 0..63
            const int ci  = c & 31;
            const int g   = row >> 4, rr = row & 15;
            const char* gp = gI + ((size_t)(g * HID + n0 + rr) * EMB) * 2 + ci * 16;
            const uint32_t lo = lds_base + (uint32_t)(RSLAB + row * IPITCH + ci * 16);
            asm volatile("global_load_async_to_lds_b128 %0, %1, off"
                         :: "v"(lo), "v"(gp) : "memory");
        }
        asm volatile("s_wait_asynccnt 0x0" ::: "memory");
    }
    __syncthreads();

    v8f acc[4] = {};                     // i, f, g, o accumulators (f32)
    const int am = m0 + lm;              // A-matrix row = batch index for this lane

    // ---- recurrent contribution: gates += h_{t-1} @ W_hh^T  (K = HID) ----
    {
        const __bf16* arow = hbf + (size_t)am * HID;
        for (int k = 0; k < HID; k += 32) {
            ABf a;
            const __bf16* ap = arow + k + lh * 8;       // A 16x32 bf16 lane layout
            a.q[0] = *(const uint4*)(ap);
            a.q[1] = *(const uint4*)(ap + 16);
            ABf b[4];                                    // batch all 4 gate B-tiles first
#pragma unroll
            for (int g = 0; g < 4; ++g) {
                const int qi = (g * 16 + lm) * (RPITCH / 16) + (k >> 3) + lh;
                b[g].q[0] = smemQ[qi];
                b[g].q[1] = smemQ[qi + 2];
            }
#pragma unroll
            for (int g = 0; g < 4; ++g) {
                acc[g] = __builtin_amdgcn_wmma_f32_16x16x32_bf16(
                    false, a.v, false, b[g].v, (short)0, acc[g], false, false);
            }
        }
    }

    // ---- input contribution: gates += emb[x[t]] @ W_ih^T  (K = EMB) ----
    {
        const int tok = x[t * BB + am];
        const float* erow = emb + (size_t)tok * EMB;
        for (int k = 0; k < EMB; k += 32) {
            ABf a;
            const float* p0 = erow + k + lh * 8;
            const float* p1 = p0 + 16;
#pragma unroll
            for (int i = 0; i < 8; ++i) a.v[i] = (__bf16)p0[i];
#pragma unroll
            for (int i = 0; i < 8; ++i) a.v[8 + i] = (__bf16)p1[i];
            ABf b[4];
#pragma unroll
            for (int g = 0; g < 4; ++g) {
                const int qi = (RSLAB / 16) + (g * 16 + lm) * (IPITCH / 16) + (k >> 3) + lh;
                b[g].q[0] = smemQ[qi];
                b[g].q[1] = smemQ[qi + 2];
            }
#pragma unroll
            for (int g = 0; g < 4; ++g) {
                acc[g] = __builtin_amdgcn_wmma_f32_16x16x32_bf16(
                    false, a.v, false, b[g].v, (short)0, acc[g], false, false);
            }
        }
    }

    // ---- epilogue: bias, nonlinearity, cell/hidden update, packed-seq freeze mask ----
    const int col = n0 + lm;                   // D tile: N = lane%16
    const float bI = bsum[0 * HID + col];
    const float bF = bsum[1 * HID + col];
    const float bG = bsum[2 * HID + col];
    const float bO = bsum[3 * HID + col];
#pragma unroll
    for (int r = 0; r < 8; ++r) {
        const int mr = m0 + r + 8 * lh;        // D tile: M = vgpr + 8*(lane/16)
        const float gi = fast_sigmoid(acc[0][r] + bI);
        const float gf = fast_sigmoid(acc[1][r] + bF);
        const float gg = fast_tanh   (acc[2][r] + bG);
        const float go = fast_sigmoid(acc[3][r] + bO);
        const size_t idx = (size_t)mr * HID + col;
        const float cold = cf[idx];
        const float hold = hf[idx];
        const float cn = gf * cold + gi * gg;
        const float hn = go * fast_tanh(cn);
        const bool live = t < x_len[mr];
        const float cw = live ? cn : cold;
        const float hw = live ? hn : hold;
        cf[idx]  = cw;
        hf[idx]  = hw;
        hbf[idx] = (__bf16)hw;                 // bf16 copy feeds next step's A operand
    }
}

// Final FC + sigmoid: 64 outputs, negligible cost.
__global__ __launch_bounds__(64) void fc_kernel(
    const float* __restrict__ hf, const float* __restrict__ fcw,
    const float* __restrict__ fcb, float* __restrict__ out)
{
    const int b = threadIdx.x;
    float s = 0.0f;
    for (int j = 0; j < HID; ++j) s += hf[b * HID + j] * fcw[j];
    out[b] = 1.0f / (1.0f + __expf(-(s + fcb[0])));
}

extern "C" void kernel_launch(void* const* d_in, const int* in_sizes, int n_in,
                              void* d_out, int out_size, void* d_ws, size_t ws_size,
                              hipStream_t stream) {
    (void)in_sizes; (void)n_in; (void)out_size; (void)ws_size;
    const int*   x     = (const int*)  d_in[0];
    const int*   x_len = (const int*)  d_in[1];
    const float* emb   = (const float*)d_in[2];
    const float* W_ih  = (const float*)d_in[3];
    const float* W_hh  = (const float*)d_in[4];
    const float* b_ih  = (const float*)d_in[5];
    const float* b_hh  = (const float*)d_in[6];
    const float* fc_w  = (const float*)d_in[7];
    const float* fc_b  = (const float*)d_in[8];
    float* out = (float*)d_out;

    char* ws = (char*)d_ws;
    size_t o = 0;
    __bf16* WhhB = (__bf16*)(ws + o); o += (size_t)G4 * HID * 2;   // 2 MB
    __bf16* WihB = (__bf16*)(ws + o); o += (size_t)G4 * EMB * 2;   // 1 MB
    __bf16* hbf  = (__bf16*)(ws + o); o += (size_t)BB * HID * 2;   // 64 KB
    float*  hf   = (float*)(ws + o);  o += (size_t)BB * HID * 4;   // 128 KB
    float*  cf   = (float*)(ws + o);  o += (size_t)BB * HID * 4;   // 128 KB
    float*  bsum = (float*)(ws + o);  o += (size_t)G4 * 4;         // 8 KB

    prep_kernel<<<(G4 * HID + 255) / 256, 256, 0, stream>>>(
        W_ih, W_hh, b_ih, b_hh, WihB, WhhB, bsum, hbf, hf, cf);

    for (int t = 0; t < TT; ++t) {
        lstm_step_kernel<<<HID / 16, 128, 0, stream>>>(
            x, x_len, emb, WihB, WhhB, bsum, hbf, hf, cf, t);
    }

    fc_kernel<<<1, 64, 0, stream>>>(hf, fc_w, fc_b, out);
}
